// MixtureOfExperts_56032143343807
// MI455X (gfx1250) — matrix operand order repaired and checked by reading, not compiled, not run on hardware
//
#include <hip/hip_runtime.h>
#include <hip/hip_bf16.h>
#include <math.h>

// ---------------- problem constants (from reference) ----------------
constexpr int kB   = 4;
constexpr int kS   = 2048;
constexpr int kD   = 768;
constexpr int kH   = 3072;
constexpr int kE   = 8;
constexpr int kK   = 2;
constexpr int kCap = 1536;            // int(B*S/E * 1.5)
constexpr int kTok  = kB * kS;        // 8192
constexpr int kFlat = kTok * kK;      // 16384

typedef __bf16 v8bf  __attribute__((ext_vector_type(8)));
typedef __bf16 v16bf __attribute__((ext_vector_type(16)));
typedef float  v8f   __attribute__((ext_vector_type(8)));

// =====================================================================
// 1) Router: one wave32 per token. logits = x @ Wr, softmax, top-2,
//    renormalize. Butterfly reduction leaves full sums in all lanes.
// =====================================================================
__global__ void __launch_bounds__(256)
moe_router_kernel(const float* __restrict__ x, const float* __restrict__ Wr,
                  int* __restrict__ eidx, float* __restrict__ gate)
{
    int tok  = blockIdx.x * 8 + (threadIdx.x >> 5);
    int lane = threadIdx.x & 31;
    const float* xr = x + (size_t)tok * kD;

    float acc[kE];
#pragma unroll
    for (int e = 0; e < kE; ++e) acc[e] = 0.f;

    for (int d = lane; d < kD; d += 32) {
        float xv = xr[d];
#pragma unroll
        for (int e = 0; e < kE; ++e) acc[e] += xv * Wr[d * kE + e];
    }
#pragma unroll
    for (int off = 16; off > 0; off >>= 1) {
#pragma unroll
        for (int e = 0; e < kE; ++e) acc[e] += __shfl_xor(acc[e], off, 32);
    }
    if (lane == 0) {
        float m = acc[0];
#pragma unroll
        for (int e = 1; e < kE; ++e) m = fmaxf(m, acc[e]);
        float p[kE];
#pragma unroll
        for (int e = 0; e < kE; ++e) p[e] = __expf(acc[e] - m);
        // top-2 (softmax denominator cancels in the renormalized gates)
        int   i0 = 0; float b0 = p[0];
#pragma unroll
        for (int e = 1; e < kE; ++e) if (p[e] > b0) { b0 = p[e]; i0 = e; }
        int   i1 = -1; float b1v = -1.f;
#pragma unroll
        for (int e = 0; e < kE; ++e) if (e != i0 && p[e] > b1v) { b1v = p[e]; i1 = e; }
        float inv = 1.f / (b0 + b1v);
        eidx[tok * 2 + 0] = i0;  gate[tok * 2 + 0] = b0  * inv;
        eidx[tok * 2 + 1] = i1;  gate[tok * 2 + 1] = b1v * inv;
    }
}

// =====================================================================
// 2) Capacity assignment: deterministic blocked scan over 16384 flat
//    entries (i innermost) -> exact "first `cap` in flat order" per expert.
// =====================================================================
__global__ void __launch_bounds__(256)
moe_assign_kernel(const int* __restrict__ eidx, int* __restrict__ slot_of,
                  int* __restrict__ tok_of, int* __restrict__ counts)
{
    __shared__ int lcnt[256][kE];
    int t = threadIdx.x;
    constexpr int PER = kFlat / 256;   // 64 consecutive entries per thread

    int c[kE];
#pragma unroll
    for (int e = 0; e < kE; ++e) c[e] = 0;
    for (int j = 0; j < PER; ++j) c[eidx[t * PER + j]]++;
#pragma unroll
    for (int e = 0; e < kE; ++e) lcnt[t][e] = c[e];
    __syncthreads();

    if (t < kE) {                       // per-expert exclusive scan over chunks
        int run = 0;
        for (int i = 0; i < 256; ++i) { int v = lcnt[i][t]; lcnt[i][t] = run; run += v; }
        counts[t] = run > kCap ? kCap : run;
    }
    __syncthreads();

    int off[kE];
#pragma unroll
    for (int e = 0; e < kE; ++e) off[e] = lcnt[t][e];
    for (int j = 0; j < PER; ++j) {
        int n = t * PER + j;
        int e = eidx[n];
        int pos = off[e]++;
        if (pos < kCap) {
            slot_of[n] = pos;
            tok_of[e * kCap + pos] = n >> 1;   // token id
        } else {
            slot_of[n] = -1;                   // dropped
        }
    }
}

// =====================================================================
// 3) Gather routed tokens -> bf16 xg[E][cap][D]; zero unused slots.
//    float4 in, packed 4x bf16 (b64) out. 192 threads = kD/4 quads.
// =====================================================================
__global__ void __launch_bounds__(192)
moe_gather_kernel(const float* __restrict__ x, const int* __restrict__ tok_of,
                  const int* __restrict__ counts, __bf16* __restrict__ xg)
{
    int slot = blockIdx.x;              // 0 .. E*cap-1
    int e = slot / kCap, s = slot - e * kCap;
    int i = threadIdx.x;                // quad index, kD/4 == 192
    union { uint2 u; __bf16 h[4]; } o;
    if (s < counts[e]) {
        float4 v = *(reinterpret_cast<const float4*>(x + (size_t)tok_of[slot] * kD) + i);
        o.h[0] = (__bf16)v.x; o.h[1] = (__bf16)v.y;
        o.h[2] = (__bf16)v.z; o.h[3] = (__bf16)v.w;
    } else {
        o.h[0] = o.h[1] = o.h[2] = o.h[3] = (__bf16)0.f;
    }
    *(reinterpret_cast<uint2*>(xg + (size_t)slot * kD) + i) = o.u;
}

// =====================================================================
// WMMA GEMM core: 128 threads = 4 waves (2x2), block tile 128x128,
// wave tile 64x64 (4x4 of 16x16), K step 32, bf16 A/B, f32 accum.
//
// Nothing is loop-carried across the barrier (no spill pressure):
//  - Store phase: A loads (4x b128) issued first, B loads (8x b128,
//    one base + immediate offsets) second; A ds_store needs only
//    loadcnt<=8 (in-order returns), B cvt+store waits loadcnt 0.
//  - Both tiles primed by global_prefetch_b8 during the prior compute
//    phase (A: kt+1, 128 row-chunks; B: kt+2, 128 cache lines = 16KB).
//  - Compute-phase live set ~178 VGPRs: acc(128) + B frags(32) + A frag(8).
//
// bf16 A frag (16x32): lane<16 holds row m=lane, K[0..8)+K[16..24);
//                      lane>=16 holds row m=lane-16, K[8..16)+K[24..32).
// bf16 B frag (32x16): lane<16 holds col n=lane, K[0..8)+K[16..24);
//                      lane>=16 holds col n=lane-16, K[8..16)+K[24..32).
// C/D (16x16 f32): vgpr r -> (M=r, N=lane) lanes 0-15, (M=r+8) lanes 16-31.
// =====================================================================
constexpr int BM = 128, BN = 128, BK = 32, BKP = 48; // BKP*2 = 96B row (16B mult.)

__device__ __forceinline__ v16bf load_frag(const __bf16* p0, const __bf16* p1)
{
    v8bf lo = *reinterpret_cast<const v8bf*>(p0);
    v8bf hi = *reinterpret_cast<const v8bf*>(p1);
    return __builtin_shufflevector(lo, hi, 0,1,2,3,4,5,6,7,8,9,10,11,12,13,14,15);
}

// KDIM: reduction length (A row length, B row count). LDN: B/out row length.
// GELU=true -> bias+gelu, bf16 store (GEMM1); false -> bias, f32 store (GEMM2).
template<int KDIM, int LDN, bool GELU>
__global__ void __launch_bounds__(128, 1)
moe_ffn_gemm(const __bf16* __restrict__ Abase, const float* __restrict__ Bwbase,
             const float* __restrict__ bias, __bf16* __restrict__ out_bf,
             float* __restrict__ out_f)
{
    int e  = blockIdx.z;
    int mb = blockIdx.x * BM;
    int nb = blockIdx.y * BN;
    const __bf16* A  = Abase  + (size_t)e * kCap * KDIM;
    const float*  Bw = Bwbase + (size_t)e * KDIM * LDN;

    __shared__ __bf16 As[BM][BKP];
    __shared__ __bf16 Bs[BN][BKP];

    int t = threadIdx.x, lane = t & 31, wv = t >> 5;
    int wm = (wv >> 1) * 64, wn = (wv & 1) * 64;
    int l15 = lane & 15, hi = lane >> 4;
    int tcol = t & 31;                   // B n-quad (invariant across i)
    int trow = t >> 5;                   // B base k-pair

    v8f acc[4][4];
    v8f zero = {0.f,0.f,0.f,0.f,0.f,0.f,0.f,0.f};
#pragma unroll
    for (int mi = 0; mi < 4; ++mi)
#pragma unroll
        for (int ni = 0; ni < 4; ++ni) acc[mi][ni] = zero;

    constexpr int NK = KDIM / BK;
    const __bf16* arow = A + (size_t)(mb + t) * KDIM;   // this thread's A row

    for (int kt = 0; kt < NK; ++kt) {
        int k0 = kt * BK;
        __syncthreads();                 // all waves done reading LDS (prev iter)

        // ---------- store phase (all staging transient) ----------
        {
            // A loads first (retire first; prefetched last iteration)
            const uint4* asrc = reinterpret_cast<const uint4*>(arow + k0);
            uint4 a0 = asrc[0], a1 = asrc[1], a2 = asrc[2], a3 = asrc[3];
            // B loads: one base + immediate row offsets (LDN constexpr)
            const float* bbase = Bw + (size_t)(k0 + 2 * trow) * LDN + nb + tcol * 4;
            float4 b0[4], b1[4];
#pragma unroll
            for (int i = 0; i < 4; ++i) {
                b0[i] = *reinterpret_cast<const float4*>(bbase + (size_t)(8 * i) * LDN);
                b1[i] = *reinterpret_cast<const float4*>(bbase + (size_t)(8 * i + 1) * LDN);
            }
            // A -> LDS (waits loadcnt<=8: A done, B still in flight)
            uint4* ad = reinterpret_cast<uint4*>(&As[t][0]);
            ad[0] = a0; ad[1] = a1; ad[2] = a2; ad[3] = a3;
            // B cvt f32->bf16 (packed), transposed store: one LDS base + imms
            __bf16* bsb = &Bs[tcol * 4][2 * trow];
#pragma unroll
            for (int i = 0; i < 4; ++i) {
                const float* f0 = reinterpret_cast<const float*>(&b0[i]);
                const float* f1 = reinterpret_cast<const float*>(&b1[i]);
#pragma unroll
                for (int j = 0; j < 4; ++j) {
                    union { unsigned u; __bf16 h[2]; } pk;
                    pk.h[0] = (__bf16)f0[j]; pk.h[1] = (__bf16)f1[j];
                    *reinterpret_cast<unsigned*>(bsb + j * BKP + 8 * i) = pk.u;
                }
            }
        }
        __syncthreads();                 // tile visible to all waves

        // ---------- compute phase ----------
        if (kt + 1 < NK)                 // prime WGP$ for A tile kt+1
            __builtin_prefetch(arow + (kt + 1) * BK, 0, 1);
        if (kt + 2 < NK) {               // prime WGP$ for B tile kt+2:
            int k2 = (kt + 2) * BK;      // 32 rows x 512B = 128 lines, 1/thread
            __builtin_prefetch(Bw + (size_t)(k2 + (t >> 2)) * LDN + nb + (t & 3) * 32, 0, 1);
        }

        v16bf bfr[4];                    // keep only B frags live through WMMAs
#pragma unroll
        for (int ni = 0; ni < 4; ++ni) {
            const __bf16* pb = &Bs[wn + ni * 16 + l15][hi * 8];
            bfr[ni] = load_frag(pb, pb + 16);
        }
#pragma unroll
        for (int mi = 0; mi < 4; ++mi) {
            const __bf16* pa = &As[wm + mi * 16 + l15][hi * 8];
            v16bf af = load_frag(pa, pa + 16);
#pragma unroll
            for (int ni = 0; ni < 4; ++ni)
                acc[mi][ni] = __builtin_amdgcn_wmma_f32_16x16x32_bf16(
                    false, af, false, bfr[ni], (short)0, acc[mi][ni], false, false);
        }
    }

    // ---------------- epilogue ----------------
#pragma unroll
    for (int ni = 0; ni < 4; ++ni) {
        int gn = nb + wn + ni * 16 + l15;
        float bv = bias[e * LDN + gn];
#pragma unroll
        for (int mi = 0; mi < 4; ++mi) {
            int gm0 = mb + wm + mi * 16 + hi * 8;
#pragma unroll
            for (int r = 0; r < 8; ++r) {
                float v = acc[mi][ni][r] + bv;
                size_t idx = ((size_t)e * kCap + gm0 + r) * LDN + gn;
                if constexpr (GELU) {
                    v = 0.5f * v * (1.0f + erff(v * 0.70710678118654752f));
                    out_bf[idx] = (__bf16)v;
                } else {
                    out_f[idx] = v;
                }
            }
        }
    }
}

// =====================================================================
// 4) Combine (deterministic gather; overwrites every output element).
//    float4 path, 192 threads = kD/4 quads per token.
// =====================================================================
__global__ void __launch_bounds__(192)
moe_combine_kernel(const int* __restrict__ eidx, const int* __restrict__ slot_of,
                   const float* __restrict__ gate, const float* __restrict__ eout,
                   float* __restrict__ out)
{
    int tok = blockIdx.x;
    int n0 = tok * 2, n1 = n0 + 1;
    int s0 = slot_of[n0], s1 = slot_of[n1];
    int e0 = eidx[n0],    e1 = eidx[n1];
    float g0 = gate[n0],  g1 = gate[n1];
    const float4* r0 = reinterpret_cast<const float4*>(
        eout + ((size_t)e0 * kCap + (s0 < 0 ? 0 : s0)) * kD);
    const float4* r1 = reinterpret_cast<const float4*>(
        eout + ((size_t)e1 * kCap + (s1 < 0 ? 0 : s1)) * kD);
    int i = threadIdx.x;                 // kD/4 == 192
    float4 s = {0.f, 0.f, 0.f, 0.f};
    if (s0 >= 0) { float4 a = r0[i]; s.x += g0*a.x; s.y += g0*a.y; s.z += g0*a.z; s.w += g0*a.w; }
    if (s1 >= 0) { float4 a = r1[i]; s.x += g1*a.x; s.y += g1*a.y; s.z += g1*a.z; s.w += g1*a.w; }
    *(reinterpret_cast<float4*>(out + (size_t)tok * kD) + i) = s;
}

// =====================================================================
extern "C" void kernel_launch(void* const* d_in, const int* in_sizes, int n_in,
                              void* d_out, int out_size, void* d_ws, size_t ws_size,
                              hipStream_t stream)
{
    const float* x  = (const float*)d_in[0];
    const float* Wr = (const float*)d_in[1];
    const float* w1 = (const float*)d_in[2];
    const float* b1 = (const float*)d_in[3];
    const float* w2 = (const float*)d_in[4];
    const float* b2 = (const float*)d_in[5];
    float* out = (float*)d_out;

    char* p = (char*)d_ws;
    auto carve = [&](size_t bytes) { char* r = p; p += (bytes + 255) & ~(size_t)255; return r; };
    int*    eidx    = (int*)   carve(sizeof(int)   * kFlat);
    int*    slot_of = (int*)   carve(sizeof(int)   * kFlat);
    int*    tok_of  = (int*)   carve(sizeof(int)   * kE * kCap);
    int*    counts  = (int*)   carve(sizeof(int)   * kE);
    float*  gate    = (float*) carve(sizeof(float) * kFlat);
    __bf16* xg      = (__bf16*)carve(sizeof(__bf16) * (size_t)kE * kCap * kD);
    __bf16* hid     = (__bf16*)carve(sizeof(__bf16) * (size_t)kE * kCap * kH);
    float*  eout    = (float*) carve(sizeof(float)  * (size_t)kE * kCap * kD);

    moe_router_kernel <<<kTok / 8, 256, 0, stream>>>(x, Wr, eidx, gate);
    moe_assign_kernel <<<1, 256, 0, stream>>>(eidx, slot_of, tok_of, counts);
    moe_gather_kernel <<<kE * kCap, 192, 0, stream>>>(x, tok_of, counts, xg);
    moe_ffn_gemm<kD, kH, true>
        <<<dim3(kCap / BM, kH / BN, kE), 128, 0, stream>>>(xg, w1, b1, hid, nullptr);
    moe_ffn_gemm<kH, kD, false>
        <<<dim3(kCap / BM, kD / BN, kE), 128, 0, stream>>>(hid, w2, b2, nullptr, eout);
    moe_combine_kernel<<<kTok, 192, 0, stream>>>(eidx, slot_of, gate, eout, out);
}
